// LSTMModel_76252849373550
// MI455X (gfx1250) — matrix operand order repaired and checked by reading, not compiled, not run on hardware
//
#include <hip/hip_runtime.h>
#include <hip/hip_bf16.h>

typedef __attribute__((ext_vector_type(16))) _Float16 v16h;
typedef __attribute__((ext_vector_type(8)))  _Float16 v8h;
typedef __attribute__((ext_vector_type(8)))  float    v8f;

#define BATCH 512
#define TSEQ  1024
#define HID   64
#define NCLS  76
#define MTILE 16      // batch rows per workgroup
#define NTHREADS 512  // 16 waves of 32

// LDS layout (bytes):
//  W0   : [256][64]  f16  = 32768   (w_hh0, row-major [gate][k])
//  W1   : [256][128] f16  = 65536   ([w_ih1 | w_hh1], row-major [gate][k])
//  gates: [16][256]  f32  = 16384
//  H0   : [16][64]   f16  =  2048   (layer-0 h_prev)
//  H1   : [16][128]  f16  =  4096   ([h1_t | layer-1 h_prev])
//  xlds : [16][128]  f32  =  8192   (x chunk, 128 timesteps)
#define SMEM_BYTES (32768 + 65536 + 16384 + 2048 + 4096 + 8192)

// Fast activations: v_exp_f32 + v_rcp_f32, no IEEE division chains.
__device__ __forceinline__ float fast_sigmoid(float x) {
  return __builtin_amdgcn_rcpf(1.0f + __expf(-x));
}
__device__ __forceinline__ float fast_tanh(float x) {
  // tanh(x) = 1 - 2/(exp(2x)+1)
  return 1.0f - 2.0f * __builtin_amdgcn_rcpf(1.0f + __expf(2.0f * x));
}

// A fragment: 16x32 f16 (MxK). Lane L: m = L&15; halves 0-7 hold
// K = kbase + (L>=16 ? 8 : 0) + [0..7], halves 8-15 hold K+16 of same.
__device__ __forceinline__ v16h load_a_frag(const _Float16* h, int stride,
                                            int lane, int kbase) {
  int m   = lane & 15;
  int khi = (lane >> 4) & 1;
  const _Float16* p = h + m * stride + kbase + khi * 8;
  v8h lo = *(const v8h*)(p);
  v8h hi = *(const v8h*)(p + 16);
  v16h a;
#pragma unroll
  for (int i = 0; i < 8; ++i) { a[i] = lo[i]; a[i + 8] = hi[i]; }
  return a;
}

// B fragment: 32x16 f16 (KxN). Lane L: n = ntile*16 + (L&15);
// halves 0-7 hold K = kbase + (L>=16 ? 8 : 0) + [0..7], halves 8-15 hold K+16 of same.
__device__ __forceinline__ v16h load_b_frag(const _Float16* w, int stride,
                                            int lane, int ntile, int kbase) {
  int n  = ntile * 16 + (lane & 15);
  int k0 = kbase + ((lane >> 4) & 1) * 8;
  const _Float16* p = w + n * stride + k0;
  v8h lo = *(const v8h*)(p);
  v8h hi = *(const v8h*)(p + 16);
  v16h b;
#pragma unroll
  for (int i = 0; i < 8; ++i) { b[i] = lo[i]; b[i + 8] = hi[i]; }
  return b;
}

// One wave computes a 16x16 gate tile with the column bias pre-folded into
// the accumulator (bias is per-column == per-lane constant in C layout).
__device__ __forceinline__ void gemm_tile(const _Float16* A, int astride,
                                          const _Float16* W, int wstride,
                                          int kchunks, int lane, int ntile,
                                          float colbias, float* gates) {
  v8f acc;
#pragma unroll
  for (int r = 0; r < 8; ++r) acc[r] = colbias;
#pragma unroll
  for (int kc = 0; kc < kchunks; ++kc) {
    v16h a = load_a_frag(A, astride, lane, kc * 32);
    v16h b = load_b_frag(W, wstride, lane, ntile, kc * 32);
    acc = __builtin_amdgcn_wmma_f32_16x16x32_f16(false, a, false, b,
                                                 (short)0, acc, false, false);
  }
  // C/D layout: VGPR r -> row r + 8*(lane>=16), col = lane&15 within tile.
  int col   = ntile * 16 + (lane & 15);
  int rbase = ((lane >> 4) & 1) * 8;
#pragma unroll
  for (int r = 0; r < 8; ++r) gates[(rbase + r) * 256 + col] = acc[r];
}

__global__ __launch_bounds__(NTHREADS)
void lstm_fused_kernel(const float* __restrict__ x,
                       const float* __restrict__ w_ih0, const float* __restrict__ w_hh0,
                       const float* __restrict__ b_ih0, const float* __restrict__ b_hh0,
                       const float* __restrict__ w_ih1, const float* __restrict__ w_hh1,
                       const float* __restrict__ b_ih1, const float* __restrict__ b_hh1,
                       const float* __restrict__ w_fc,  const float* __restrict__ b_fc,
                       float* __restrict__ out) {
  extern __shared__ char smem_raw[];
  _Float16* W0    = (_Float16*)smem_raw;            // [256][64]
  _Float16* W1    = W0 + 256 * 64;                  // [256][128]
  float*    gates = (float*)(W1 + 256 * 128);       // [16][256]
  _Float16* H0    = (_Float16*)(gates + 16 * 256);  // [16][64]
  _Float16* H1    = H0 + 16 * 64;                   // [16][128]
  float*    xlds  = (float*)(H1 + 16 * 128);        // [16][128]

  const int tid  = threadIdx.x;
  const int lane = tid & 31;
  const int wid  = tid >> 5;           // wave id == gate n-tile, 0..15
  const int b0   = blockIdx.x * MTILE; // batch base for this WG

  // ---- stage weights into LDS as f16 (resident for the whole scan) ----
  for (int i = tid; i < 256 * 64; i += NTHREADS) W0[i] = (_Float16)w_hh0[i];
  for (int i = tid; i < 256 * 128; i += NTHREADS) {
    int n = i >> 7, k = i & 127;
    float v = (k < 64) ? w_ih1[n * 64 + k] : w_hh1[n * 64 + (k - 64)];
    W1[i] = (_Float16)v;
  }
  for (int i = tid; i < 16 * 64; i += NTHREADS)  H0[i] = (_Float16)0.0f;
  for (int i = tid; i < 16 * 128; i += NTHREADS) H1[i] = (_Float16)0.0f;

  // Per-lane gate-column bias, folded into the WMMA accumulator init.
  const int gcol = wid * 16 + (lane & 15);          // 0..255
  const float bias0c = b_ih0[gcol] + b_hh0[gcol];
  const float bias1c = b_ih1[gcol] + b_hh1[gcol];

  // ---- per-thread elementwise ownership: 1 batch row, 2 hidden units ----
  const int ew_b  = (tid * 2) >> 6;  // 0..15
  const int ew_h0 = (tid * 2) & 63;  // even hidden index
  float c0s[2] = {0.f, 0.f}, c1s[2] = {0.f, 0.f};
  float wih0v[2][4];
#pragma unroll
  for (int u = 0; u < 2; ++u) {
    int hd = ew_h0 + u;
#pragma unroll
    for (int q = 0; q < 4; ++q)       // gate order i,f,g,o at q*64+hd
      wih0v[u][q] = w_ih0[q * 64 + hd];   // input size == 1
  }
  __syncthreads();

  for (int t = 0; t < TSEQ; ++t) {
    if ((t & 127) == 0) {
      // prefetch 128 timesteps of x for the 16 batch rows (coalesced)
      int row = tid >> 5;
      int c4  = (tid & 31) * 4;
      const float* xg = x + (size_t)(b0 + row) * TSEQ + t + c4;
      *(float4*)&xlds[row * 128 + c4] = *(const float4*)xg;
    }
    // ---------- layer 0: gates = h_prev0 @ w_hh0^T + bias  (K=64) ----------
    gemm_tile(H0, 64, W0, 64, 2, lane, wid, bias0c, gates);
    __syncthreads();
    {
      float xv = xlds[ew_b * 128 + (t & 127)];
      const float* gr = gates + ew_b * 256;
#pragma unroll
      for (int u = 0; u < 2; ++u) {
        int hd = ew_h0 + u;
        float gi = gr[hd]       + xv * wih0v[u][0];
        float gf = gr[64 + hd]  + xv * wih0v[u][1];
        float gg = gr[128 + hd] + xv * wih0v[u][2];
        float go = gr[192 + hd] + xv * wih0v[u][3];
        float ii = fast_sigmoid(gi), ff = fast_sigmoid(gf);
        float g  = fast_tanh(gg),    oo = fast_sigmoid(go);
        c0s[u] = ff * c0s[u] + ii * g;
        float h = oo * fast_tanh(c0s[u]);
        _Float16 hh = (_Float16)h;
        H0[ew_b * 64 + hd]  = hh;   // h_prev0 for next step
        H1[ew_b * 128 + hd] = hh;   // h1_t input to layer 1 this step
      }
    }
    __syncthreads();
    // ---- layer 1: gates = [h1_t | h_prev1] @ [w_ih1;w_hh1]^T + bias (K=128) ----
    gemm_tile(H1, 128, W1, 128, 4, lane, wid, bias1c, gates);
    __syncthreads();
    {
      const float* gr = gates + ew_b * 256;
#pragma unroll
      for (int u = 0; u < 2; ++u) {
        int hd = ew_h0 + u;
        float ii = fast_sigmoid(gr[hd]);
        float ff = fast_sigmoid(gr[64 + hd]);
        float g  = fast_tanh(gr[128 + hd]);
        float oo = fast_sigmoid(gr[192 + hd]);
        c1s[u] = ff * c1s[u] + ii * g;
        float h = oo * fast_tanh(c1s[u]);
        H1[ew_b * 128 + 64 + hd] = (_Float16)h;  // h_prev1 for next step
      }
    }
    __syncthreads();
  }

  // ---------- final FC: out[b][c] = h2_last[b] . w_fc[c] + b_fc[c] ----------
  for (int j = tid; j < MTILE * NCLS; j += NTHREADS) {
    int b = j / NCLS, cls = j - b * NCLS;
    const _Float16* hrow = H1 + b * 128 + 64;
    const float*    wrow = w_fc + cls * 64;
    float s = b_fc[cls];
#pragma unroll
    for (int k = 0; k < 64; ++k) s += (float)hrow[k] * wrow[k];
    out[(size_t)(b0 + b) * NCLS + cls] = s;
  }
}

extern "C" void kernel_launch(void* const* d_in, const int* in_sizes, int n_in,
                              void* d_out, int out_size, void* d_ws, size_t ws_size,
                              hipStream_t stream) {
  (void)in_sizes; (void)n_in; (void)out_size; (void)d_ws; (void)ws_size;
  const float* xx    = (const float*)d_in[0];
  const float* wih0  = (const float*)d_in[1];
  const float* whh0  = (const float*)d_in[2];
  const float* bih0  = (const float*)d_in[3];
  const float* bhh0  = (const float*)d_in[4];
  const float* wih1  = (const float*)d_in[5];
  const float* whh1  = (const float*)d_in[6];
  const float* bih1  = (const float*)d_in[7];
  const float* bhh1  = (const float*)d_in[8];
  const float* wfc   = (const float*)d_in[9];
  const float* bfc   = (const float*)d_in[10];
  float* out = (float*)d_out;

  dim3 grid(BATCH / MTILE);   // 32 workgroups, each owns 16 batch rows
  dim3 block(NTHREADS);       // 16 waves
  lstm_fused_kernel<<<grid, block, SMEM_BYTES, stream>>>(
      xx, wih0, whh0, bih0, bhh0, wih1, whh1, bih1, bhh1, wfc, bfc, out);
}
